// VRNN_85813446574515
// MI455X (gfx1250) — matrix operand order repaired and checked by reading, not compiled, not run on hardware
//
#include <hip/hip_runtime.h>
#include <hip/hip_bf16.h>

// ---------------------------------------------------------------------------
// VRNN forward on gfx1250 (MI455X): bf16 WMMA GEMMs + f32 accumulation.
// ---------------------------------------------------------------------------

typedef __attribute__((ext_vector_type(16))) __bf16 v16bf;
typedef __attribute__((ext_vector_type(8)))  __bf16 bf16x8;
typedef __attribute__((ext_vector_type(8)))  float  v8f;

#define VR_B     2048
#define VR_T     64
#define VR_X     2
#define VR_Z     64
#define VR_H     256
#define VR_R     512
#define VR_LOG2PI 1.8378770664093453f

// ---------------------------------------------------------------------------
// Generic GEMM:  C[M,N] = act( A[M,K] @ W[N,K]^T + bias[N] )
// A, W bf16; bias f32; optional f32 and/or bf16 outputs.
// Block: 256 threads = 8 waves; wave tile 32x32 (4 WMMA accs); block tile 128x64.
// Requires M%128==0, N%64==0, K%32==0 (true for all shapes used here).
// ---------------------------------------------------------------------------
template<int ACT>
__global__ __launch_bounds__(256)
void vr_gemm_bf16(const __bf16* __restrict__ A, const __bf16* __restrict__ W,
                  const float* __restrict__ bias,
                  float* __restrict__ outF, __bf16* __restrict__ outB,
                  int M, int N, int K)
{
    const int lane = threadIdx.x & 31;
    const int wave = threadIdx.x >> 5;
    const int wm   = wave & 3;                  // 0..3  (M direction)
    const int wn   = wave >> 2;                 // 0..1  (N direction)
    const int m0   = blockIdx.x * 128 + wm * 32;
    const int n0   = blockIdx.y * 64  + wn * 32;

    v8f acc00 = {}, acc01 = {}, acc10 = {}, acc11 = {};

    // A fragment (16x32 bf16): lane<16 -> K {k..k+15}; lane>=16 -> +16
    const int aRow0 = m0 + (lane & 15);
    const int ksel  = (lane >> 4) ? 16 : 0;
    // B fragment (32x16 bf16): lane holds col n, 16 contiguous K starting at kb
    const int bCol0 = n0 + (lane & 15);
    const int kb    = (lane >> 4) ? 16 : 0;

    const __bf16* aP0 = A + (size_t)aRow0 * K + ksel;
    const __bf16* aP1 = aP0 + (size_t)16 * K;
    const __bf16* bP0 = W + (size_t)bCol0 * K + kb;
    const __bf16* bP1 = bP0 + (size_t)16 * K;

    for (int k = 0; k < K; k += 32) {
        if (k + 32 < K) {
            // WGP-scope (near-cache) prefetch of the next K slab:
            // locality 3 -> keep in all cache levels (global_prefetch_b8, WGP scope).
            __builtin_prefetch(aP0 + k + 32, 0, 3);
            __builtin_prefetch(bP0 + k + 32, 0, 3);
        }
        v16bf a0, a1;
        bf16x8 lo, hi;
        lo = *(const bf16x8*)(aP0 + k);
        hi = *(const bf16x8*)(aP0 + k + 8);
#pragma unroll
        for (int i = 0; i < 8; ++i) { a0[i] = lo[i]; a0[8 + i] = hi[i]; }
        lo = *(const bf16x8*)(aP1 + k);
        hi = *(const bf16x8*)(aP1 + k + 8);
#pragma unroll
        for (int i = 0; i < 8; ++i) { a1[i] = lo[i]; a1[8 + i] = hi[i]; }

        v16bf b0 = *(const v16bf*)(bP0 + k);
        v16bf b1 = *(const v16bf*)(bP1 + k);

        acc00 = __builtin_amdgcn_wmma_f32_16x16x32_bf16(false, a0, false, b0, (short)0, acc00, false, false);
        acc01 = __builtin_amdgcn_wmma_f32_16x16x32_bf16(false, a0, false, b1, (short)0, acc01, false, false);
        acc10 = __builtin_amdgcn_wmma_f32_16x16x32_bf16(false, a1, false, b0, (short)0, acc10, false, false);
        acc11 = __builtin_amdgcn_wmma_f32_16x16x32_bf16(false, a1, false, b1, (short)0, acc11, false, false);
    }

    // C/D layout: lane<16 -> M=r, N=lane ; lane>=16 -> M=8+r, N=lane-16
    const int col   = lane & 15;
    const int rBase = (lane >> 4) * 8;

    auto store_tile = [&](const v8f& acc, int mOff, int nOff) {
        const int n  = n0 + nOff + col;
        const float bv = bias ? bias[n] : 0.0f;
        const int mB = m0 + mOff + rBase;
#pragma unroll
        for (int r = 0; r < 8; ++r) {
            float v = acc[r] + bv;
            if (ACT) v = (v > 0.0f) ? v : 0.01f * v;   // leaky_relu(0.01)
            const size_t idx = (size_t)(mB + r) * N + n;
            if (outF) outF[idx] = v;
            if (outB) outB[idx] = (__bf16)v;
        }
    };
    store_tile(acc00,  0,  0);
    store_tile(acc01,  0, 16);
    store_tile(acc10, 16,  0);
    store_tile(acc11, 16, 16);
}

// ---------------------------------------------------------------------------
// Small elementwise / degenerate-shape kernels
// ---------------------------------------------------------------------------

__global__ void vr_cvt_bf16(const float* __restrict__ s, __bf16* __restrict__ d, int n)
{
    int i = blockIdx.x * blockDim.x + threadIdx.x;
    if (i < n) d[i] = (__bf16)s[i];
}

// phi_x layer 1: K=2 -> pure VALU. out = lrelu(W1 @ x + b1), bf16 out.
__global__ void vr_phix1(const float* __restrict__ xt, const float* __restrict__ w,
                         const float* __restrict__ b, __bf16* __restrict__ out)
{
    int idx = blockIdx.x * blockDim.x + threadIdx.x;      // B*256 threads
    int bi = idx >> 8, n = idx & 255;
    float v = w[n * 2] * xt[bi * 2] + w[n * 2 + 1] * xt[bi * 2 + 1] + b[n];
    v = (v > 0.0f) ? v : 0.01f * v;
    out[idx] = (__bf16)v;
}

// concat along feature dim: out[b, 0:wa]=a, out[b, wa:wa+wb]=bsrc
__global__ void vr_concat2(const __bf16* __restrict__ a, int wa,
                           const __bf16* __restrict__ bsrc, int wb,
                           __bf16* __restrict__ out)
{
    int idx = blockIdx.x * blockDim.x + threadIdx.x;      // B*(wa+wb) threads
    int w = wa + wb;
    int row = idx / w, j = idx % w;
    out[idx] = (j < wa) ? a[(size_t)row * wa + j] : bsrc[(size_t)row * wb + (j - wa)];
}

// z = eps * exp(0.5*logvar) + mean  (bf16 out for phi_z GEMM)
__global__ void vr_reparam(const float* __restrict__ eps, const float* __restrict__ mean,
                           const float* __restrict__ logvar, __bf16* __restrict__ z, int n)
{
    int i = blockIdx.x * blockDim.x + threadIdx.x;
    if (i < n) z[i] = (__bf16)(eps[i] * __expf(0.5f * logvar[i]) + mean[i]);
}

// KLD accumulation: one thread per sample, LDS tree reduce, atomicAdd.
__global__ void vr_kld(const float* __restrict__ em, const float* __restrict__ el,
                       const float* __restrict__ pm, const float* __restrict__ pl,
                       float* __restrict__ kld_out)
{
    int b = blockIdx.x * blockDim.x + threadIdx.x;        // B threads
    float x1 = 0.f, x2 = 0.f, x3 = 0.f;
    const size_t base = (size_t)b * VR_Z;
    for (int j = 0; j < VR_Z; ++j) {
        float e = el[base + j], p = pl[base + j];
        float dm = em[base + j] - pm[base + j];
        x1 += p - e;
        x2 += __expf(e - p);
        x3 += dm * dm * __expf(-p);
    }
    float v = 0.5f * (x1 - (float)VR_Z + x2 + x3) * (1.0f / (float)VR_B);
    __shared__ float red[256];
    red[threadIdx.x] = v;
    __syncthreads();
    for (int s = 128; s > 0; s >>= 1) {
        if (threadIdx.x < s) red[threadIdx.x] += red[threadIdx.x + s];
        __syncthreads();
    }
    if (threadIdx.x == 0) atomicAdd(kld_out, red[0]);
}

// Decoder heads (N=2 each) fused with Gaussian NLL accumulation.
__global__ void vr_dec_head_nll(const __bf16* __restrict__ dec_h,
                                const float* __restrict__ wm, const float* __restrict__ bm,
                                const float* __restrict__ wl, const float* __restrict__ bl,
                                const float* __restrict__ xt, float* __restrict__ nll_out)
{
    int b = blockIdx.x * blockDim.x + threadIdx.x;        // B threads
    float m0 = bm[0], m1 = bm[1], l0 = bl[0], l1 = bl[1];
    const __bf16* h = dec_h + (size_t)b * VR_H;
    for (int k = 0; k < VR_H; ++k) {
        float hv = (float)h[k];
        m0 += wm[k] * hv;  m1 += wm[VR_H + k] * hv;
        l0 += wl[k] * hv;  l1 += wl[VR_H + k] * hv;
    }
    float x0 = xt[b * 2], x1 = xt[b * 2 + 1];
    float n1 = (x0 - m0) * (x0 - m0) * __expf(-l0) + (x1 - m1) * (x1 - m1) * __expf(-l1);
    float v = 0.5f * (n1 + 2.0f * VR_LOG2PI + l0 + l1) * (1.0f / (float)VR_B);
    __shared__ float red[256];
    red[threadIdx.x] = v;
    __syncthreads();
    for (int s = 128; s > 0; s >>= 1) {
        if (threadIdx.x < s) red[threadIdx.x] += red[threadIdx.x + s];
        __syncthreads();
    }
    if (threadIdx.x == 0) atomicAdd(nll_out, red[0]);
}

// GRU gate math (PyTorch r,z,n order). gx/gh already include biases from GEMM.
__global__ void vr_gru_gate(const float* __restrict__ gx, const float* __restrict__ gh,
                            const float* __restrict__ hcur,
                            float* __restrict__ hnext, __bf16* __restrict__ hnext_bf)
{
    int i = blockIdx.x * blockDim.x + threadIdx.x;        // B*512 threads
    int b = i >> 9, j = i & 511;
    const float* gxr = gx + (size_t)b * 1536;
    const float* ghr = gh + (size_t)b * 1536;
    float r  = 1.0f / (1.0f + __expf(-(gxr[j]        + ghr[j])));
    float zg = 1.0f / (1.0f + __expf(-(gxr[512 + j]  + ghr[512 + j])));
    float n  = tanhf(gxr[1024 + j] + r * ghr[1024 + j]);
    float h  = hcur[i];
    float hn = (1.0f - zg) * n + zg * h;
    hnext[i]    = hn;
    hnext_bf[i] = (__bf16)hn;
}

// ---------------------------------------------------------------------------
// Host-side orchestration
// ---------------------------------------------------------------------------

static inline void launch_gemm(const __bf16* A, const __bf16* W, const float* bias,
                               float* oF, __bf16* oB, int M, int N, int K, bool act,
                               hipStream_t s)
{
    dim3 g(M / 128, N / 64);
    if (act) vr_gemm_bf16<1><<<g, 256, 0, s>>>(A, W, bias, oF, oB, M, N, K);
    else     vr_gemm_bf16<0><<<g, 256, 0, s>>>(A, W, bias, oF, oB, M, N, K);
}

extern "C" void kernel_launch(void* const* d_in, const int* in_sizes, int n_in,
                              void* d_out, int out_size, void* d_ws, size_t ws_size,
                              hipStream_t stream)
{
    (void)in_sizes; (void)n_in; (void)out_size; (void)ws_size;
    const int B = VR_B;

    // ---- inputs (setup_inputs dict order; params flattened) ----
    const float* x          = (const float*)d_in[0];   // [64, B, 2]
    const float* phi_x_w1   = (const float*)d_in[1];
    const float* phi_x_b1   = (const float*)d_in[2];
    const float* phi_x_w2   = (const float*)d_in[3];
    const float* phi_x_b2   = (const float*)d_in[4];
    const float* phi_z_w1   = (const float*)d_in[5];
    const float* phi_z_b1   = (const float*)d_in[6];
    const float* phi_z_w2   = (const float*)d_in[7];
    const float* phi_z_b2   = (const float*)d_in[8];
    const float* enc_w1     = (const float*)d_in[9];
    const float* enc_b1     = (const float*)d_in[10];
    const float* enc_w2     = (const float*)d_in[11];
    const float* enc_b2     = (const float*)d_in[12];
    const float* prior_w1   = (const float*)d_in[13];
    const float* prior_b1   = (const float*)d_in[14];
    const float* prior_w2   = (const float*)d_in[15];
    const float* prior_b2   = (const float*)d_in[16];
    const float* dec_w1     = (const float*)d_in[17];
    const float* dec_b1     = (const float*)d_in[18];
    const float* dec_w2     = (const float*)d_in[19];
    const float* dec_b2     = (const float*)d_in[20];
    const float* enc_mean_w = (const float*)d_in[21];
    const float* enc_mean_b = (const float*)d_in[22];
    const float* enc_lv_w   = (const float*)d_in[23];
    const float* enc_lv_b   = (const float*)d_in[24];
    const float* pri_mean_w = (const float*)d_in[25];
    const float* pri_mean_b = (const float*)d_in[26];
    const float* pri_lv_w   = (const float*)d_in[27];
    const float* pri_lv_b   = (const float*)d_in[28];
    const float* dec_mean_w = (const float*)d_in[29];
    const float* dec_mean_b = (const float*)d_in[30];
    const float* dec_lv_w   = (const float*)d_in[31];
    const float* dec_lv_b   = (const float*)d_in[32];
    const float* gru_wih    = (const float*)d_in[33];  // [1536, 512]
    const float* gru_bih    = (const float*)d_in[34];
    const float* gru_whh    = (const float*)d_in[35];  // [1536, 512]
    const float* gru_bhh    = (const float*)d_in[36];
    const float* eps        = (const float*)d_in[37];  // [63, B, 64]

    float* outF = (float*)d_out;                       // [kld, nll, h(B*512)]

    // ---- workspace bump allocator (256B aligned) ----
    char*  base = (char*)d_ws;
    size_t off  = 0;
    auto alloc = [&](size_t bytes) -> void* {
        void* p = base + off;
        off += (bytes + 255) & ~(size_t)255;
        return p;
    };
    auto allocB = [&](size_t n) { return (__bf16*)alloc(n * sizeof(__bf16)); };
    auto allocF = [&](size_t n) { return (float*)alloc(n * sizeof(float)); };

    // bf16 weight mirrors
    __bf16* wb_phix2  = allocB(256 * 256);
    __bf16* wb_phiz1  = allocB(256 * 64);
    __bf16* wb_phiz2  = allocB(256 * 256);
    __bf16* wb_enc1   = allocB(256 * 768);
    __bf16* wb_enc2   = allocB(256 * 256);
    __bf16* wb_pri1   = allocB(256 * 512);
    __bf16* wb_pri2   = allocB(256 * 256);
    __bf16* wb_dec1   = allocB(256 * 768);
    __bf16* wb_dec2   = allocB(256 * 256);
    __bf16* wb_encm   = allocB(64 * 256);
    __bf16* wb_encl   = allocB(64 * 256);
    __bf16* wb_prim   = allocB(64 * 256);
    __bf16* wb_pril   = allocB(64 * 256);
    __bf16* wb_wih    = allocB(1536 * 512);
    __bf16* wb_whh    = allocB(1536 * 512);

    // activations
    __bf16* tmp256 = allocB((size_t)B * 256);
    __bf16* phi_x  = allocB((size_t)B * 256);
    __bf16* prior_h = allocB((size_t)B * 256);
    __bf16* enc_h  = allocB((size_t)B * 256);
    __bf16* phi_z  = allocB((size_t)B * 256);
    __bf16* dec_h  = allocB((size_t)B * 256);
    __bf16* zbf    = allocB((size_t)B * 64);
    __bf16* cat768 = allocB((size_t)B * 768);
    __bf16* gi     = allocB((size_t)B * 512);
    __bf16* hB[2]  = { allocB((size_t)B * 512), allocB((size_t)B * 512) };

    float* enc_mean = allocF((size_t)B * 64);
    float* enc_lv   = allocF((size_t)B * 64);
    float* pri_mean = allocF((size_t)B * 64);
    float* pri_lv   = allocF((size_t)B * 64);
    float* hF[2]    = { allocF((size_t)B * 512), allocF((size_t)B * 512) };
    float* gx       = allocF((size_t)B * 1536);
    float* gh       = allocF((size_t)B * 1536);

    // ---- one-time-per-call setup ----
    hipMemsetAsync(d_out, 0, 2 * sizeof(float), stream);
    hipMemsetAsync(hF[0], 0, (size_t)B * 512 * sizeof(float), stream);
    hipMemsetAsync(hB[0], 0, (size_t)B * 512 * sizeof(__bf16), stream);

    auto cvt = [&](const float* s, __bf16* d, int n) {
        vr_cvt_bf16<<<(n + 255) / 256, 256, 0, stream>>>(s, d, n);
    };
    cvt(phi_x_w2, wb_phix2, 256 * 256);
    cvt(phi_z_w1, wb_phiz1, 256 * 64);
    cvt(phi_z_w2, wb_phiz2, 256 * 256);
    cvt(enc_w1,   wb_enc1,  256 * 768);
    cvt(enc_w2,   wb_enc2,  256 * 256);
    cvt(prior_w1, wb_pri1,  256 * 512);
    cvt(prior_w2, wb_pri2,  256 * 256);
    cvt(dec_w1,   wb_dec1,  256 * 768);
    cvt(dec_w2,   wb_dec2,  256 * 256);
    cvt(enc_mean_w, wb_encm, 64 * 256);
    cvt(enc_lv_w,   wb_encl, 64 * 256);
    cvt(pri_mean_w, wb_prim, 64 * 256);
    cvt(pri_lv_w,   wb_pril, 64 * 256);
    cvt(gru_wih,  wb_wih, 1536 * 512);
    cvt(gru_whh,  wb_whh, 1536 * 512);

    // ---- sequential scan over t = 1..63 ----
    for (int t = 1; t < VR_T; ++t) {
        const int cur = (t - 1) & 1;
        const int nxt = t & 1;
        const float* xt  = x + (size_t)t * B * 2;
        const float* ept = eps + (size_t)(t - 1) * B * 64;

        // phi_x = mlp(xt)
        vr_phix1<<<(B * 256) / 256, 256, 0, stream>>>(xt, phi_x_w1, phi_x_b1, tmp256);
        launch_gemm(tmp256, wb_phix2, phi_x_b2, nullptr, phi_x, B, 256, 256, true, stream);

        // prior = mlp(ht); prior heads
        launch_gemm(hB[cur], wb_pri1, prior_b1, nullptr, tmp256, B, 256, 512, true, stream);
        launch_gemm(tmp256, wb_pri2, prior_b2, nullptr, prior_h, B, 256, 256, true, stream);
        launch_gemm(prior_h, wb_prim, pri_mean_b, pri_mean, nullptr, B, 64, 256, false, stream);
        launch_gemm(prior_h, wb_pril, pri_lv_b,   pri_lv,   nullptr, B, 64, 256, false, stream);

        // enc = mlp([phi_x, ht]); enc heads
        vr_concat2<<<(B * 768) / 256, 256, 0, stream>>>(phi_x, 256, hB[cur], 512, cat768);
        launch_gemm(cat768, wb_enc1, enc_b1, nullptr, tmp256, B, 256, 768, true, stream);
        launch_gemm(tmp256, wb_enc2, enc_b2, nullptr, enc_h, B, 256, 256, true, stream);
        launch_gemm(enc_h, wb_encm, enc_mean_b, enc_mean, nullptr, B, 64, 256, false, stream);
        launch_gemm(enc_h, wb_encl, enc_lv_b,   enc_lv,   nullptr, B, 64, 256, false, stream);

        // KLD accumulation
        vr_kld<<<B / 256, 256, 0, stream>>>(enc_mean, enc_lv, pri_mean, pri_lv, outF + 0);

        // z = eps*exp(0.5*logvar)+mean; phi_z = mlp(z)
        vr_reparam<<<(B * 64) / 256, 256, 0, stream>>>(ept, enc_mean, enc_lv, zbf, B * 64);
        launch_gemm(zbf,    wb_phiz1, phi_z_b1, nullptr, tmp256, B, 256, 64, true, stream);
        launch_gemm(tmp256, wb_phiz2, phi_z_b2, nullptr, phi_z, B, 256, 256, true, stream);

        // dec = mlp([phi_z, ht]); dec heads + NLL
        vr_concat2<<<(B * 768) / 256, 256, 0, stream>>>(phi_z, 256, hB[cur], 512, cat768);
        launch_gemm(cat768, wb_dec1, dec_b1, nullptr, tmp256, B, 256, 768, true, stream);
        launch_gemm(tmp256, wb_dec2, dec_b2, nullptr, dec_h, B, 256, 256, true, stream);
        vr_dec_head_nll<<<B / 256, 256, 0, stream>>>(dec_h, dec_mean_w, dec_mean_b,
                                                     dec_lv_w, dec_lv_b, xt, outF + 1);

        // GRU: gx = [phi_x|phi_z] @ wih^T + bih ; gh = ht @ whh^T + bhh
        vr_concat2<<<(B * 512) / 256, 256, 0, stream>>>(phi_x, 256, phi_z, 256, gi);
        launch_gemm(gi,      wb_wih, gru_bih, gx, nullptr, B, 1536, 512, false, stream);
        launch_gemm(hB[cur], wb_whh, gru_bhh, gh, nullptr, B, 1536, 512, false, stream);
        vr_gru_gate<<<(B * 512) / 256, 256, 0, stream>>>(gx, gh, hF[cur], hF[nxt], hB[nxt]);
    }

    // final hidden state -> d_out[2:]
    hipMemcpyAsync(outF + 2, hF[(VR_T - 1) & 1], (size_t)B * 512 * sizeof(float),
                   hipMemcpyDeviceToDevice, stream);
}
